// PocketGNN_67705864454312
// MI455X (gfx1250) — compile-verified, matrix operand order checked by reading, not run on hardware
//
#include <hip/hip_runtime.h>
#include <hip/hip_bf16.h>

typedef __attribute__((ext_vector_type(16))) _Float16 v16h;
typedef __attribute__((ext_vector_type(8)))  _Float16 v8h;
typedef __attribute__((ext_vector_type(8)))  float    v8f;
typedef __attribute__((ext_vector_type(4)))  float    v4f;

#define NNODES 10000
#define NEDGES 160000
#define NGRAPH 64
#define HID    256

// ---------------------------------------------------------------------------
// Repack f32 weight W[K][Nout] (row-major) into f16 WMMA B-fragment order:
// frag[((nt*KT + kt)*32 + lane)*16 + j] = W[k][n]
//   n = nt*16 + (lane&15)
//   k = kt*32 + (j>>3)*16 + (lane>>4)*8 + (j&7)
// so a lane's 16 halves are contiguous (two b128 loads).
// ---------------------------------------------------------------------------
__global__ __launch_bounds__(256)
void repack_weight_kernel(const float* __restrict__ W, _Float16* __restrict__ F,
                          int K, int Nout) {
    int idx = blockIdx.x * 256 + threadIdx.x;
    if (idx >= K * Nout) return;
    int j       = idx & 15;
    int L       = (idx >> 4) & 31;
    int tileIdx = idx >> 9;
    int numKT   = K >> 5;
    int kt = tileIdx % numKT;
    int nt = tileIdx / numKT;
    int n  = nt * 16 + (L & 15);
    int k  = kt * 32 + ((j >> 3) << 4) + ((L >> 4) << 3) + (j & 7);
    F[idx] = (_Float16)W[(size_t)k * Nout + n];
}

// ---------------------------------------------------------------------------
// Fused EdgeConv, 32 edges (2 WMMA M-tiles) per workgroup so every B-fragment
// load feeds two WMMAs: build m=[x_i, x_j-x_i] (f16, LDS), GEMM1 (2*CIN->256)
// + bias + relu, GEMM2 (256->256) + bias, float atomic-max scatter into
// out[dst].  out pre-zeroed => the reference's trailing relu is implicit.
// ---------------------------------------------------------------------------
template <int CIN>
__global__ __launch_bounds__(256)
void edgeconv_kernel(const float* __restrict__ xin,
                     const int* __restrict__ srcIdx,
                     const int* __restrict__ dstIdx,
                     const _Float16* __restrict__ w1f, const float* __restrict__ b1,
                     const _Float16* __restrict__ w2f, const float* __restrict__ b2,
                     float* __restrict__ out) {
    constexpr int K1  = 2 * CIN;
    constexpr int KT1 = K1 / 32;
    constexpr int KT2 = HID / 32;

    __shared__ __align__(16) _Float16 mL[32 * K1];
    __shared__ __align__(16) _Float16 h1[32 * HID];
    __shared__ int dstS[32];

    const int tile  = blockIdx.x;       // 32-edge tile
    const int ebase = tile * 32;
    const int t     = threadIdx.x;

    if (t < 32) dstS[t] = dstIdx[ebase + t];

    // -------- phase 1: gather m = [x_i, x_j - x_i] into LDS as f16 --------
    {
        const int e    = t >> 3;        // edge row 0..31
        const int part = t & 7;
        constexpr int span = K1 / 8;    // 32 or 64 consecutive k per thread
        const int kb = part * span;
        const int si = srcIdx[ebase + e];
        const int di = dstIdx[ebase + e];
        const float* xi = xin + (size_t)di * CIN;
        const float* xj = xin + (size_t)si * CIN;
        _Float16* mrow = &mL[e * K1];
        if (kb < CIN) {
            #pragma unroll
            for (int k = 0; k < span; k += 4) {
                v4f v = *(const v4f*)(xi + kb + k);
                mrow[kb + k + 0] = (_Float16)v[0];
                mrow[kb + k + 1] = (_Float16)v[1];
                mrow[kb + k + 2] = (_Float16)v[2];
                mrow[kb + k + 3] = (_Float16)v[3];
            }
        } else {
            const int c = kb - CIN;
            #pragma unroll
            for (int k = 0; k < span; k += 4) {
                v4f a = *(const v4f*)(xj + c + k);
                v4f b = *(const v4f*)(xi + c + k);
                mrow[kb + k + 0] = (_Float16)(a[0] - b[0]);
                mrow[kb + k + 1] = (_Float16)(a[1] - b[1]);
                mrow[kb + k + 2] = (_Float16)(a[2] - b[2]);
                mrow[kb + k + 3] = (_Float16)(a[3] - b[3]);
            }
        }
    }
    __syncthreads();

    const int lane = t & 31;
    const int half = lane >> 4;
    const int l16  = lane & 15;
    const int wv   = t >> 5;            // wave 0..7
    const int nt0  = wv * 2;
    const int nt1  = wv * 2 + 1;

    // -------- phase 2: GEMM1 (m @ W1) + bias + relu -> h1 (f16, LDS) ------
    v8f acc00 = {0, 0, 0, 0, 0, 0, 0, 0};   // [mtile0][nt0]
    v8f acc01 = {0, 0, 0, 0, 0, 0, 0, 0};   // [mtile0][nt1]
    v8f acc10 = {0, 0, 0, 0, 0, 0, 0, 0};   // [mtile1][nt0]
    v8f acc11 = {0, 0, 0, 0, 0, 0, 0, 0};   // [mtile1][nt1]
    {
        const _Float16* mrow0 = &mL[l16 * K1];
        const _Float16* mrow1 = &mL[(16 + l16) * K1];
        #pragma unroll
        for (int kt = 0; kt < KT1; ++kt) {
            v8h lo0 = *(const v8h*)(mrow0 + kt * 32 + half * 8);
            v8h hi0 = *(const v8h*)(mrow0 + kt * 32 + 16 + half * 8);
            v8h lo1 = *(const v8h*)(mrow1 + kt * 32 + half * 8);
            v8h hi1 = *(const v8h*)(mrow1 + kt * 32 + 16 + half * 8);
            v16h a0 = __builtin_shufflevector(lo0, hi0, 0, 1, 2, 3, 4, 5, 6, 7,
                                                        8, 9, 10, 11, 12, 13, 14, 15);
            v16h a1 = __builtin_shufflevector(lo1, hi1, 0, 1, 2, 3, 4, 5, 6, 7,
                                                        8, 9, 10, 11, 12, 13, 14, 15);
            v16h bA = *(const v16h*)(w1f + ((size_t)(nt0 * KT1 + kt) * 32 + lane) * 16);
            v16h bB = *(const v16h*)(w1f + ((size_t)(nt1 * KT1 + kt) * 32 + lane) * 16);
            acc00 = __builtin_amdgcn_wmma_f32_16x16x32_f16(false, a0, false, bA,
                                                           (short)0, acc00, false, false);
            acc10 = __builtin_amdgcn_wmma_f32_16x16x32_f16(false, a1, false, bA,
                                                           (short)0, acc10, false, false);
            acc01 = __builtin_amdgcn_wmma_f32_16x16x32_f16(false, a0, false, bB,
                                                           (short)0, acc01, false, false);
            acc11 = __builtin_amdgcn_wmma_f32_16x16x32_f16(false, a1, false, bB,
                                                           (short)0, acc11, false, false);
        }
        const float bias0 = b1[nt0 * 16 + l16];
        const float bias1 = b1[nt1 * 16 + l16];
        #pragma unroll
        for (int r = 0; r < 8; ++r) {
            const int m0 = r + 8 * half;
            const int m1 = 16 + m0;
            float v00 = acc00[r] + bias0; v00 = v00 > 0.f ? v00 : 0.f;
            float v01 = acc01[r] + bias1; v01 = v01 > 0.f ? v01 : 0.f;
            float v10 = acc10[r] + bias0; v10 = v10 > 0.f ? v10 : 0.f;
            float v11 = acc11[r] + bias1; v11 = v11 > 0.f ? v11 : 0.f;
            h1[m0 * HID + nt0 * 16 + l16] = (_Float16)v00;
            h1[m0 * HID + nt1 * 16 + l16] = (_Float16)v01;
            h1[m1 * HID + nt0 * 16 + l16] = (_Float16)v10;
            h1[m1 * HID + nt1 * 16 + l16] = (_Float16)v11;
        }
    }
    __syncthreads();

    // -------- phase 3: GEMM2 (h1 @ W2) + bias ------------------------------
    v8f d00 = {0, 0, 0, 0, 0, 0, 0, 0};
    v8f d01 = {0, 0, 0, 0, 0, 0, 0, 0};
    v8f d10 = {0, 0, 0, 0, 0, 0, 0, 0};
    v8f d11 = {0, 0, 0, 0, 0, 0, 0, 0};
    {
        const _Float16* hrow0 = &h1[l16 * HID];
        const _Float16* hrow1 = &h1[(16 + l16) * HID];
        #pragma unroll
        for (int kt = 0; kt < KT2; ++kt) {
            v8h lo0 = *(const v8h*)(hrow0 + kt * 32 + half * 8);
            v8h hi0 = *(const v8h*)(hrow0 + kt * 32 + 16 + half * 8);
            v8h lo1 = *(const v8h*)(hrow1 + kt * 32 + half * 8);
            v8h hi1 = *(const v8h*)(hrow1 + kt * 32 + 16 + half * 8);
            v16h a0 = __builtin_shufflevector(lo0, hi0, 0, 1, 2, 3, 4, 5, 6, 7,
                                                        8, 9, 10, 11, 12, 13, 14, 15);
            v16h a1 = __builtin_shufflevector(lo1, hi1, 0, 1, 2, 3, 4, 5, 6, 7,
                                                        8, 9, 10, 11, 12, 13, 14, 15);
            v16h bA = *(const v16h*)(w2f + ((size_t)(nt0 * KT2 + kt) * 32 + lane) * 16);
            v16h bB = *(const v16h*)(w2f + ((size_t)(nt1 * KT2 + kt) * 32 + lane) * 16);
            d00 = __builtin_amdgcn_wmma_f32_16x16x32_f16(false, a0, false, bA,
                                                         (short)0, d00, false, false);
            d10 = __builtin_amdgcn_wmma_f32_16x16x32_f16(false, a1, false, bA,
                                                         (short)0, d10, false, false);
            d01 = __builtin_amdgcn_wmma_f32_16x16x32_f16(false, a0, false, bB,
                                                         (short)0, d01, false, false);
            d11 = __builtin_amdgcn_wmma_f32_16x16x32_f16(false, a1, false, bB,
                                                         (short)0, d11, false, false);
        }
    }

    // -------- phase 4: bias + atomic-max scatter --------------------------
    {
        const float bias0 = b2[nt0 * 16 + l16];
        const float bias1 = b2[nt1 * 16 + l16];
        #pragma unroll
        for (int r = 0; r < 8; ++r) {
            const int m0 = r + 8 * half;
            const int m1 = 16 + m0;
            const int node0 = dstS[m0];
            const int node1 = dstS[m1];
            float* p00 = out + (size_t)node0 * HID + nt0 * 16 + l16;
            float* p01 = out + (size_t)node0 * HID + nt1 * 16 + l16;
            float* p10 = out + (size_t)node1 * HID + nt0 * 16 + l16;
            float* p11 = out + (size_t)node1 * HID + nt1 * 16 + l16;
            __hip_atomic_fetch_max(p00, d00[r] + bias0,
                                   __ATOMIC_RELAXED, __HIP_MEMORY_SCOPE_AGENT);
            __hip_atomic_fetch_max(p01, d01[r] + bias1,
                                   __ATOMIC_RELAXED, __HIP_MEMORY_SCOPE_AGENT);
            __hip_atomic_fetch_max(p10, d10[r] + bias0,
                                   __ATOMIC_RELAXED, __HIP_MEMORY_SCOPE_AGENT);
            __hip_atomic_fetch_max(p11, d11[r] + bias1,
                                   __ATOMIC_RELAXED, __HIP_MEMORY_SCOPE_AGENT);
        }
    }
}

// ---------------------------------------------------------------------------
// Per-graph mean/max pooling via atomics (x >= 0 post-relu => 0-init max ok)
// ---------------------------------------------------------------------------
__global__ __launch_bounds__(256)
void pool_kernel(const float* __restrict__ x, const int* __restrict__ batch,
                 float* __restrict__ sums, float* __restrict__ maxs,
                 float* __restrict__ counts) {
    int idx = blockIdx.x * 256 + threadIdx.x;
    if (idx >= NNODES * HID) return;
    int node = idx >> 8;
    int col  = idx & (HID - 1);
    int g    = batch[node];
    float v  = x[idx];
    atomicAdd(&sums[g * HID + col], v);
    __hip_atomic_fetch_max(&maxs[g * HID + col], v,
                           __ATOMIC_RELAXED, __HIP_MEMORY_SCOPE_AGENT);
    if (col == 0) atomicAdd(&counts[g], 1.0f);
}

// ---------------------------------------------------------------------------
// Head MLP: [G,512] -> 256 relu -> 64 relu -> 1   (tiny; one block per graph)
// ---------------------------------------------------------------------------
__global__ __launch_bounds__(256)
void mlp_kernel(const float* __restrict__ sums, const float* __restrict__ maxs,
                const float* __restrict__ counts,
                const float* __restrict__ w1, const float* __restrict__ bb1,
                const float* __restrict__ w2, const float* __restrict__ bb2,
                const float* __restrict__ w3, const float* __restrict__ bb3,
                float* __restrict__ out) {
    __shared__ float h[2 * HID];
    __shared__ float o1[HID];
    __shared__ float o2[64];
    const int g = blockIdx.x;
    const int t = threadIdx.x;
    const float inv = 1.0f / fmaxf(counts[g], 1.0f);
    h[t]       = sums[g * HID + t] * inv;
    h[HID + t] = maxs[g * HID + t];
    __syncthreads();
    float a1 = bb1[t];
    for (int k = 0; k < 2 * HID; ++k) a1 += h[k] * w1[k * HID + t];
    o1[t] = fmaxf(a1, 0.0f);
    __syncthreads();
    if (t < 64) {
        float a2 = bb2[t];
        for (int k = 0; k < HID; ++k) a2 += o1[k] * w2[k * 64 + t];
        o2[t] = fmaxf(a2, 0.0f);
    }
    __syncthreads();
    if (t == 0) {
        float a3 = bb3[0];
        for (int k = 0; k < 64; ++k) a3 += o2[k] * w3[k];
        out[g] = a3;
    }
}

// ---------------------------------------------------------------------------
extern "C" void kernel_launch(void* const* d_in, const int* in_sizes, int n_in,
                              void* d_out, int out_size, void* d_ws, size_t ws_size,
                              hipStream_t stream) {
    const float* x     = (const float*)d_in[0];
    const int*   eidx  = (const int*)d_in[1];
    const int*   batch = (const int*)d_in[2];
    const float* cw1[3] = {(const float*)d_in[3], (const float*)d_in[7],  (const float*)d_in[11]};
    const float* cb1[3] = {(const float*)d_in[4], (const float*)d_in[8],  (const float*)d_in[12]};
    const float* cw2[3] = {(const float*)d_in[5], (const float*)d_in[9],  (const float*)d_in[13]};
    const float* cb2[3] = {(const float*)d_in[6], (const float*)d_in[10], (const float*)d_in[14]};
    const float* clw1 = (const float*)d_in[15];
    const float* clb1 = (const float*)d_in[16];
    const float* clw2 = (const float*)d_in[17];
    const float* clb2 = (const float*)d_in[18];
    const float* clw3 = (const float*)d_in[19];
    const float* clb3 = (const float*)d_in[20];

    const int* srcIdx = eidx;
    const int* dstIdx = eidx + NEDGES;

    // ---- workspace carve-up ----
    char* ws = (char*)d_ws;
    const size_t bufBytes = (size_t)NNODES * HID * sizeof(float);
    float* bufA = (float*)ws;              ws += bufBytes;
    float* bufB = (float*)ws;              ws += bufBytes;
    const int K1s[3] = {256, 512, 512};
    _Float16* fW1[3];
    _Float16* fW2[3];
    for (int l = 0; l < 3; ++l) {
        fW1[l] = (_Float16*)ws;            ws += (size_t)K1s[l] * HID * sizeof(_Float16);
        fW2[l] = (_Float16*)ws;            ws += (size_t)HID * HID * sizeof(_Float16);
    }
    float* sums   = (float*)ws;            ws += (size_t)NGRAPH * HID * sizeof(float);
    float* maxs   = (float*)ws;            ws += (size_t)NGRAPH * HID * sizeof(float);
    float* counts = (float*)ws;            ws += (size_t)NGRAPH * sizeof(float);

    // ---- repack all weights into WMMA f16 fragment layout ----
    for (int l = 0; l < 3; ++l) {
        int n1 = K1s[l] * HID;
        repack_weight_kernel<<<n1 / 256, 256, 0, stream>>>(cw1[l], fW1[l], K1s[l], HID);
        int n2 = HID * HID;
        repack_weight_kernel<<<n2 / 256, 256, 0, stream>>>(cw2[l], fW2[l], HID, HID);
    }

    const int nTiles = NEDGES / 32;   // 32 edges per workgroup

    // ---- layer 0: x(128) -> bufA ----
    hipMemsetAsync(bufA, 0, bufBytes, stream);
    edgeconv_kernel<128><<<nTiles, 256, 0, stream>>>(x, srcIdx, dstIdx,
                                                     fW1[0], cb1[0], fW2[0], cb2[0], bufA);
    // ---- layer 1: bufA(256) -> bufB ----
    hipMemsetAsync(bufB, 0, bufBytes, stream);
    edgeconv_kernel<256><<<nTiles, 256, 0, stream>>>(bufA, srcIdx, dstIdx,
                                                     fW1[1], cb1[1], fW2[1], cb2[1], bufB);
    // ---- layer 2: bufB(256) -> bufA ----
    hipMemsetAsync(bufA, 0, bufBytes, stream);
    edgeconv_kernel<256><<<nTiles, 256, 0, stream>>>(bufB, srcIdx, dstIdx,
                                                     fW1[2], cb1[2], fW2[2], cb2[2], bufA);

    // ---- pooling ----
    hipMemsetAsync(sums, 0, ((size_t)NGRAPH * HID * 2 + NGRAPH) * sizeof(float), stream);
    pool_kernel<<<(NNODES * HID + 255) / 256, 256, 0, stream>>>(bufA, batch, sums, maxs, counts);

    // ---- head MLP ----
    mlp_kernel<<<NGRAPH, 256, 0, stream>>>(sums, maxs, counts,
                                           clw1, clb1, clw2, clb2, clw3, clb3,
                                           (float*)d_out);
}
